// RNNCell_5531917877722
// MI455X (gfx1250) — matrix operand order repaired and checked by reading, not compiled, not run on hardware
//
#include <hip/hip_runtime.h>

// ---- problem dims (fixed by the reference) ----
#define B_   64
#define T_   512
#define F_   256
#define H_   512
#define G4_  2048   // 4*H
#define PAD_ 8      // LDS row pad (bf16 elems) -> conflict-free b128 frag reads

typedef __attribute__((ext_vector_type(16))) __bf16 v16bf;
typedef __attribute__((ext_vector_type(8)))  __bf16 v8bf;
typedef __attribute__((ext_vector_type(8)))  float  v8f;

// ---------------- helpers ----------------
__device__ __forceinline__ __bf16 f2bf(float f) {
  union { float f; unsigned u; } x; x.f = f;
  unsigned r = x.u + 0x7FFFu + ((x.u >> 16) & 1u);   // round-to-nearest-even
  unsigned short h = (unsigned short)(r >> 16);
  return __builtin_bit_cast(__bf16, h);
}

__device__ __forceinline__ float sigm(float x) { return 1.0f / (1.0f + __expf(-x)); }
__device__ __forceinline__ float tanh_fast(float x) { return 2.0f / (1.0f + __expf(-2.0f * x)) - 1.0f; }

// Async global->LDS B128 copy (CDNA5 GLOBAL_LOAD_ASYNC_TO_LDS_B128, ASYNCcnt),
// with a portable fallback that still stages through LDS (ds_store_b128).
#if defined(__gfx1250__) && __has_builtin(__builtin_amdgcn_global_load_async_to_lds_b128)
#define HAVE_ASYNC_LDS 1
#else
#define HAVE_ASYNC_LDS 0
#endif

typedef int v4i_async __attribute__((vector_size(16)));

__device__ __forceinline__ void async_copy_b128(const __bf16* g, __bf16* l) {
#if HAVE_ASYNC_LDS
  __builtin_amdgcn_global_load_async_to_lds_b128(
      (__attribute__((address_space(1))) v4i_async*)g,
      (__attribute__((address_space(3))) v4i_async*)l,
      0, 0);
#else
  *(v8bf*)l = *(const v8bf*)g;
#endif
}

__device__ __forceinline__ void wait_async() {
#if HAVE_ASYNC_LDS
  asm volatile("s_wait_asynccnt 0x0" ::: "memory");
#endif
}

// A-matrix fragment (16x32 bf16, M x K) from an LDS row:
//   v0..3: lanes0-15 K=k0+0..7,  lanes16-31 K=k0+8..15   -> 16B at row + k0 + half*8
//   v4..7: lanes0-15 K=k0+16..23, lanes16-31 K=k0+24..31 -> 16B at row + k0 + 16 + half*8
__device__ __forceinline__ v16bf load_frag_a(const __bf16* row, int k0, int half) {
  union { v16bf v; v8bf h[2]; } u;
  u.h[0] = *(const v8bf*)(row + k0 + half * 8);
  u.h[1] = *(const v8bf*)(row + k0 + 16 + half * 8);
  return u.v;
}

// B-matrix fragment (32x16 bf16, K x N), lane n = lane%16, same K order as A:
//   v0..3: K=k0+half*8+0..7, v4..7: K=k0+16+half*8+0..7
__device__ __forceinline__ v16bf load_frag_b(const __bf16* __restrict__ row, int k0, int half) {
  union { v16bf v; v8bf h[2]; } u;
  u.h[0] = *(const v8bf*)(row + k0 + half * 8);
  u.h[1] = *(const v8bf*)(row + k0 + 16 + half * 8);
  return u.v;
}

// accumulate 4 gate tiles over one K segment; A from LDS, weights streamed from global
__device__ __forceinline__ void gemm_part(const __bf16* arow,
                                          const __bf16* const* __restrict__ wr,
                                          int K, int half, v8f* acc) {
  for (int k0 = 0; k0 < K; k0 += 32) {
    v16bf a = load_frag_a(arow, k0, half);
#pragma unroll
    for (int g = 0; g < 4; ++g) {
      v16bf b = load_frag_b(wr[g], k0, half);
      __builtin_prefetch(wr[g] + k0 + 64, 0, 0);
      acc[g] = __builtin_amdgcn_wmma_f32_16x16x32_bf16(
          /*neg_a=*/false, a, /*neg_b=*/false, b,
          /*c_mod=*/(short)0, acc[g], /*reuse_a=*/false, /*reuse_b=*/false);
    }
  }
}

// ---------------- prep kernels ----------------
__global__ void k_f32_to_bf16(const float* __restrict__ s, __bf16* __restrict__ d, int n) {
  for (int i = blockIdx.x * blockDim.x + threadIdx.x; i < n; i += gridDim.x * blockDim.x)
    d[i] = f2bf(s[i]);
}

__global__ void k_bias_sum(const float* __restrict__ a, const float* __restrict__ b,
                           float* __restrict__ d, int n) {
  int i = blockIdx.x * blockDim.x + threadIdx.x;
  if (i < n) d[i] = a[i] + b[i];
}

__global__ void k_zero_b32(unsigned* __restrict__ p, int n) {
  for (int i = blockIdx.x * blockDim.x + threadIdx.x; i < n; i += gridDim.x * blockDim.x)
    p[i] = 0u;
}

// ---------------- fused LSTM step (one layer, one timestep) ----------------
// gates = [x_t | h_prev] @ [W_ih ; W_hh]^T + bias ; pointwise update of c, h.
// Grid: (8 col-blocks of 64 h-cols, 4 row-tiles of 16 batch rows), 128 thr = 4 waves.
// Wave w handles h-cols [blk*64 + w*16, +16), with 4 accumulators (i,f,g,o).
// Activation panels (x-part, h-part) are staged once per block into LDS via async copies.
__global__ __launch_bounds__(128) void k_lstm_step(
    const __bf16* __restrict__ Ax, long sAx, int Kx,       // x-part operand, row stride, width
    const __bf16* __restrict__ Wih,                        // [4H, Kx] row-major, bf16
    const __bf16* __restrict__ hprev,                      // [B, H] bf16
    const __bf16* __restrict__ Whh,                        // [4H, H] row-major, bf16
    const float*  __restrict__ bias,                       // [4H] (b_ih + b_hh)
    float*        __restrict__ c,                          // [B, H] f32 state (in/out)
    __bf16*       __restrict__ hnext,                      // [B, H] bf16 (out)
    float*        __restrict__ out, long sOut)             // optional f32 out rows (layer 1)
{
  __shared__ __bf16 sX[16 * (H_ + PAD_)];   // x-part panel (sized for Kx<=512)
  __shared__ __bf16 sH[16 * (H_ + PAD_)];   // h-part panel

  const int tid  = threadIdx.x;
  const int lane = tid & 31;
  const int wave = tid >> 5;
  const int half = lane >> 4;
  const int nl   = lane & 15;
  const int mt   = blockIdx.y;                          // batch tile 0..3
  const int n    = blockIdx.x * 64 + wave * 16 + nl;    // hidden column 0..511

  // --- stage the 16-row activation panels into LDS (async B128 copies) ---
  const int sxStride = Kx + PAD_;
  const int cprX = Kx >> 3;                             // b128 chunks per x-row
  for (int cix = tid; cix < 16 * cprX; cix += 128) {
    const int row = cix / cprX;
    const int col = (cix - row * cprX) << 3;
    async_copy_b128(Ax + (long)(mt * 16 + row) * sAx + col, &sX[row * sxStride + col]);
  }
  for (int cih = tid; cih < 16 * (H_ >> 3); cih += 128) {
    const int row = cih >> 6;
    const int col = (cih & 63) << 3;
    async_copy_b128(hprev + (long)(mt * 16 + row) * H_ + col, &sH[row * (H_ + PAD_) + col]);
  }
  wait_async();
  __syncthreads();

  // --- GEMM: 4 gate tiles accumulated in registers ---
  v8f acc[4] = { v8f{}, v8f{}, v8f{}, v8f{} };
  const __bf16* wr[4];

#pragma unroll
  for (int g = 0; g < 4; ++g) wr[g] = Wih + (long)(g * H_ + n) * Kx;
  gemm_part(&sX[nl * sxStride], wr, Kx, half, acc);

#pragma unroll
  for (int g = 0; g < 4; ++g) wr[g] = Whh + (long)(g * H_ + n) * H_;
  gemm_part(&sH[nl * (H_ + PAD_)], wr, H_, half, acc);

  // --- pointwise gate fusion; C/D layout: VGPR r holds row M = 8*half + r, col = n ---
  const float bi = bias[0 * H_ + n];
  const float bf = bias[1 * H_ + n];
  const float bg = bias[2 * H_ + n];
  const float bo = bias[3 * H_ + n];

#pragma unroll
  for (int r = 0; r < 8; ++r) {
    const int m = mt * 16 + half * 8 + r;        // batch row
    const float gi = sigm(acc[0][r] + bi);
    const float gf = sigm(acc[1][r] + bf);
    const float gg = tanh_fast(acc[2][r] + bg);
    const float go = sigm(acc[3][r] + bo);
    const long idx = (long)m * H_ + n;
    const float cn = gf * c[idx] + gi * gg;
    c[idx] = cn;
    const float hn = go * tanh_fast(cn);
    hnext[idx] = f2bf(hn);
    if (out) out[(long)m * sOut + n] = hn;
  }
}

// ---------------- host orchestration ----------------
extern "C" void kernel_launch(void* const* d_in, const int* in_sizes, int n_in,
                              void* d_out, int out_size, void* d_ws, size_t ws_size,
                              hipStream_t stream) {
  const float* x    = (const float*)d_in[0];
  const float* Wih0 = (const float*)d_in[1];
  const float* Whh0 = (const float*)d_in[2];
  const float* bih0 = (const float*)d_in[3];
  const float* bhh0 = (const float*)d_in[4];
  const float* Wih1 = (const float*)d_in[5];
  const float* Whh1 = (const float*)d_in[6];
  const float* bih1 = (const float*)d_in[7];
  const float* bhh1 = (const float*)d_in[8];
  float* out = (float*)d_out;

  // carve workspace (256B aligned; all sizes are multiples of 256B so carves are contiguous)
  char* p = (char*)d_ws;
  auto carve = [&](size_t bytes) -> char* {
    char* r = p;
    p += (bytes + 255) & ~(size_t)255;
    return r;
  };
  __bf16* xbf   = (__bf16*)carve((size_t)B_ * T_ * F_ * 2);     // 16 MB
  __bf16* wih0  = (__bf16*)carve((size_t)G4_ * F_ * 2);         // 1 MB
  __bf16* whh0  = (__bf16*)carve((size_t)G4_ * H_ * 2);         // 2 MB
  __bf16* wih1  = (__bf16*)carve((size_t)G4_ * H_ * 2);         // 2 MB
  __bf16* whh1  = (__bf16*)carve((size_t)G4_ * H_ * 2);         // 2 MB
  float*  bias0 = (float*)carve((size_t)G4_ * 4);
  float*  bias1 = (float*)carve((size_t)G4_ * 4);
  float*  c0    = (float*)carve((size_t)B_ * H_ * 4);           // state region start
  float*  c1    = (float*)carve((size_t)B_ * H_ * 4);
  __bf16* h0[2] = { (__bf16*)carve((size_t)B_ * H_ * 2), (__bf16*)carve((size_t)B_ * H_ * 2) };
  __bf16* h1[2] = { (__bf16*)carve((size_t)B_ * H_ * 2), (__bf16*)carve((size_t)B_ * H_ * 2) };
  (void)ws_size; (void)in_sizes; (void)n_in; (void)out_size;

  // --- prep: convert to bf16, fuse biases, zero recurrent state (every call: graph replay) ---
  k_f32_to_bf16<<<2048, 256, 0, stream>>>(x,    xbf,  B_ * T_ * F_);
  k_f32_to_bf16<<<512,  256, 0, stream>>>(Wih0, wih0, G4_ * F_);
  k_f32_to_bf16<<<512,  256, 0, stream>>>(Whh0, whh0, G4_ * H_);
  k_f32_to_bf16<<<512,  256, 0, stream>>>(Wih1, wih1, G4_ * H_);
  k_f32_to_bf16<<<512,  256, 0, stream>>>(Whh1, whh1, G4_ * H_);
  k_bias_sum<<<8, 256, 0, stream>>>(bih0, bhh0, bias0, G4_);
  k_bias_sum<<<8, 256, 0, stream>>>(bih1, bhh1, bias1, G4_);
  // zero c0,c1,h0[0],h0[1],h1[0],h1[1] in one shot (contiguous carve region)
  k_zero_b32<<<128, 256, 0, stream>>>((unsigned*)c0,
      (B_ * H_ * 4 * 2 + B_ * H_ * 2 * 4) / 4);

  const dim3 grid(8, 4);   // 8 col-blocks x 4 batch tiles = 32 WGPs
  const dim3 blk(128);     // 4 waves

  // --- recurrent sweep: interleave layer0/layer1 per timestep; h double-buffered ---
  for (int t = 0; t < T_; ++t) {
    const int pi = t & 1, po = pi ^ 1;
    // layer 0: x-part from xbf[:, t, :] (row stride T*F), writes h0[po]
    k_lstm_step<<<grid, blk, 0, stream>>>(
        xbf + (size_t)t * F_, (long)T_ * F_, F_,
        wih0, h0[pi], whh0, bias0, c0, h0[po],
        nullptr, 0L);
    // layer 1: x-part = layer0 output h0[po] (row stride H), writes h1[po] + d_out[:, t, :]
    k_lstm_step<<<grid, blk, 0, stream>>>(
        h0[po], (long)H_, H_,
        wih1, h1[pi], whh1, bias1, c1, h1[po],
        out + (size_t)t * H_, (long)T_ * H_);
  }
}
